// PlanetoidSCN_54417235641005
// MI455X (gfx1250) — matrix operand order repaired and checked by reading, not compiled, not run on hardware
//
#include <hip/hip_runtime.h>
#include <hip/hip_bf16.h>

typedef __attribute__((ext_vector_type(16))) _Float16 v16h;
typedef __attribute__((ext_vector_type(8)))  float    v8f;

#define FDIM 128

// ---------------------------------------------------------------------------
// Elementwise: binarize node features
// ---------------------------------------------------------------------------
__global__ void binarize_kernel(const float* __restrict__ X, float* __restrict__ Xb, int n) {
    int i = blockIdx.x * blockDim.x + threadIdx.x;
    if (i < n) Xb[i] = (X[i] != 0.0f) ? 1.0f : 0.0f;
}

// ---------------------------------------------------------------------------
// Weight f32 -> f16 conversion (done once per launch; 128x128 each)
// ---------------------------------------------------------------------------
__global__ void cvt_f16_kernel(const float* __restrict__ W, _Float16* __restrict__ Wh, int n) {
    int i = blockIdx.x * blockDim.x + threadIdx.x;
    if (i < n) Wh[i] = (_Float16)W[i];
}

// ---------------------------------------------------------------------------
// Edge features: X1f[e,:] = X0b[a,:] * X0b[b,:]   (one wave per edge, float4)
// ---------------------------------------------------------------------------
__global__ void edge_feat_kernel(const int* __restrict__ eidx, const float* __restrict__ Xb,
                                 float* __restrict__ X1, int n1) {
    int gid  = blockIdx.x * blockDim.x + threadIdx.x;
    int e    = gid >> 5;
    int lane = gid & 31;
    if (e >= n1) return;
    int a = eidx[2 * e + 0];
    int b = eidx[2 * e + 1];
    float4 fa = ((const float4*)(Xb + (size_t)a * FDIM))[lane];
    float4 fb = ((const float4*)(Xb + (size_t)b * FDIM))[lane];
    ((float4*)(X1 + (size_t)e * FDIM))[lane] =
        make_float4(fa.x * fb.x, fa.y * fb.y, fa.z * fb.z, fa.w * fb.w);
}

// ---------------------------------------------------------------------------
// Triangle features: X2f[t,:] = X0b[a]*X0b[b]*X0b[c]
// ---------------------------------------------------------------------------
__global__ void tri_feat_kernel(const int* __restrict__ tidx, const float* __restrict__ Xb,
                                float* __restrict__ X2, int n2) {
    int gid  = blockIdx.x * blockDim.x + threadIdx.x;
    int t    = gid >> 5;
    int lane = gid & 31;
    if (t >= n2) return;
    int a = tidx[3 * t + 0];
    int b = tidx[3 * t + 1];
    int c = tidx[3 * t + 2];
    float4 fa = ((const float4*)(Xb + (size_t)a * FDIM))[lane];
    float4 fb = ((const float4*)(Xb + (size_t)b * FDIM))[lane];
    float4 fc = ((const float4*)(Xb + (size_t)c * FDIM))[lane];
    ((float4*)(X2 + (size_t)t * FDIM))[lane] =
        make_float4(fa.x * fb.x * fc.x, fa.y * fb.y * fc.y,
                    fa.z * fb.z * fc.z, fa.w * fb.w * fc.w);
}

// ---------------------------------------------------------------------------
// COO SpMM: Y[rows[e],:] += vals[e] * X[cols[e],:]   (one wave per nnz)
// float4 gather + 4 f32 atomics per lane. Y must be pre-zeroed / pre-filled.
// ---------------------------------------------------------------------------
__global__ void spmm_atomic_kernel(const int* __restrict__ rows, const int* __restrict__ cols,
                                   const float* __restrict__ vals, const float* __restrict__ X,
                                   float* __restrict__ Y, int nnz) {
    int gid  = blockIdx.x * blockDim.x + threadIdx.x;
    int e    = gid >> 5;
    int lane = gid & 31;
    if (e >= nnz) return;
    int   r = rows[e];
    int   c = cols[e];
    float v = vals[e];
    float4 x = ((const float4*)(X + (size_t)c * FDIM))[lane];
    float* y = Y + (size_t)r * FDIM + lane * 4;
    atomicAdd(y + 0, v * x.x);
    atomicAdd(y + 1, v * x.y);
    atomicAdd(y + 2, v * x.z);
    atomicAdd(y + 3, v * x.w);
}

// ---------------------------------------------------------------------------
// WMMA GEMM: H = [prelu](S @ W^T + b) [+= existing H if ACCUM]
//   S  : [N,128] f32 (converted to f16 on the fly)
//   Wh : [128,128] f16 row-major (OUT x F) -> B[k][n] = Wh[n*128+k]
// One wave computes a 16x128 output stripe: A fragment loaded once per
// K-step, reused across 8 column tiles (8 v8f accumulators).
// A/B fragments use contiguous K per lane half (K = 16*hi + h), same order in both (wave32).
// ---------------------------------------------------------------------------
template <bool PRELU, bool ACCUM>
__global__ __launch_bounds__(128)
void gemm128_kernel(const float* __restrict__ S, const _Float16* __restrict__ Wh,
                    const float* __restrict__ bias, const float* __restrict__ prelu_w,
                    float* __restrict__ H, int N) {
    const int lane = threadIdx.x;                                  // 0..31
    const int row0 = (blockIdx.x * blockDim.y + threadIdx.y) * 16; // wave-uniform
    if (row0 >= N) return;                                         // uniform exit, EXEC stays full
    const int m  = lane & 15;
    const int hi = lane >> 4;

    v8f c[8] = {};

    // clamped A row (only matters if N % 16 != 0; rows >= N are never stored)
    int arow = row0 + m;
    if (arow > N - 1) arow = N - 1;
    const float* srow = S + (size_t)arow * FDIM;

#pragma unroll
    for (int k0 = 0; k0 < FDIM; k0 += 32) {
        // A 16x32 f16: lanes 0-15 hold M=lane, K = k0+0..15;
        //              lanes 16-31 hold M=lane-16, K = k0+16..31
        v16h a;
#pragma unroll
        for (int h = 0; h < 16; ++h) {
            int k = k0 + hi * 16 + h;
            a[h] = (_Float16)srow[k];
        }
#pragma unroll
        for (int nt = 0; nt < 8; ++nt) {
            // B 32x16 f16: lane = N (mod 16), halves h -> K = k0 + hi*16 + h
            const _Float16* wrow = Wh + (size_t)(nt * 16 + m) * FDIM;
            v16h b;
#pragma unroll
            for (int h = 0; h < 16; ++h) b[h] = wrow[k0 + hi * 16 + h];
            c[nt] = __builtin_amdgcn_wmma_f32_16x16x32_f16(
                false, a, false, b, (short)0, c[nt], false, false);
        }
    }

    const float pw = PRELU ? prelu_w[0] : 0.0f;
#pragma unroll
    for (int nt = 0; nt < 8; ++nt) {
        const int   col = nt * 16 + m;
        const float bv  = bias[col];
#pragma unroll
        for (int r = 0; r < 8; ++r) {
            int row = row0 + r + hi * 8; // C/D layout: VGPR r -> M = r + 8*hi, N = lane%16
            if (row < N) {
                size_t idx = (size_t)row * FDIM + col;
                float  v   = c[nt][r] + bv;
                if (PRELU) v = (v >= 0.0f) ? v : pw * v;
                if (ACCUM) v += H[idx];
                H[idx] = v;
            }
        }
    }
}

// ---------------------------------------------------------------------------
// Final: out = acc / 3
// ---------------------------------------------------------------------------
__global__ void scale_out_kernel(const float* __restrict__ A, float* __restrict__ O, int n) {
    int i = blockIdx.x * blockDim.x + threadIdx.x;
    if (i < n) O[i] = A[i] * (1.0f / 3.0f);
}

// ---------------------------------------------------------------------------
static inline int cdiv(long long a, long long b) { return (int)((a + b - 1) / b); }

extern "C" void kernel_launch(void* const* d_in, const int* in_sizes, int n_in,
                              void* d_out, int out_size, void* d_ws, size_t ws_size,
                              hipStream_t stream) {
    // ---- inputs (setup_inputs dict order) ----
    const float* X0      = (const float*)d_in[0];
    const int*   eidx    = (const int*)d_in[1];
    const int*   tidx    = (const int*)d_in[2];
    const int*   L0r     = (const int*)d_in[3];
    const int*   L0c     = (const int*)d_in[4];
    const float* L0v     = (const float*)d_in[5];
    const int*   L1r     = (const int*)d_in[6];
    const int*   L1c     = (const int*)d_in[7];
    const float* L1v     = (const float*)d_in[8];
    const int*   L2r     = (const int*)d_in[9];
    const int*   L2c     = (const int*)d_in[10];
    const float* L2v     = (const float*)d_in[11];
    const int*   B1r     = (const int*)d_in[12];
    const int*   B1c     = (const int*)d_in[13];
    const float* B1v     = (const float*)d_in[14];
    const int*   B2r     = (const int*)d_in[15];
    const int*   B2c     = (const int*)d_in[16];
    const float* B2v     = (const float*)d_in[17];
    const float* Wn      = (const float*)d_in[18];
    const float* bn      = (const float*)d_in[19];
    const float* We      = (const float*)d_in[20];
    const float* be      = (const float*)d_in[21];
    const float* Wt      = (const float*)d_in[22];
    const float* bt      = (const float*)d_in[23];
    const float* Wtri    = (const float*)d_in[24];
    const float* btri    = (const float*)d_in[25];
    const float* prelu_w = (const float*)d_in[26];

    const int N0 = in_sizes[0] / FDIM;
    const int N1 = in_sizes[1] / 2;
    const int N2 = in_sizes[2] / 3;
    const int nnz0  = in_sizes[3];
    const int nnz1  = in_sizes[6];
    const int nnz2  = in_sizes[9];
    const int nnzb1 = in_sizes[12];
    const int nnzb2 = in_sizes[15];

    const size_t fN0 = (size_t)N0 * FDIM;
    const size_t fN1 = (size_t)N1 * FDIM;
    const size_t fN2 = (size_t)N2 * FDIM;

    // ---- workspace layout (f32 elements) ----
    // [0,        fN0)              X0b        (dead after L0 spmm + gathers)
    // [fN0,      fN0+fN1)          X1f        (dead after L1 spmm)
    // [0,        fN0+fN1)          H2  (alias; N0+N1 == N2 here, sized by max)
    // [fN0+fN1,  +fN2)             X2f
    // next fN2                     Sbuf (shared spmm accumulator: S0/S1/S2/SB2)
    // next fN1                     U    (H1, then H1 + T)
    // next fN0                     ACC0 (H0, then += B1@U)
    // next                         f16 weights: Wn,We,Wt,Wtri (4 * 128*128)
    float* ws    = (float*)d_ws;
    float* X0b   = ws;
    float* X1f   = ws + fN0;
    float* X2f   = ws + fN0 + fN1;
    float* Sbuf  = X2f + fN2;
    float* Ubuf  = Sbuf + fN2;
    float* ACC0  = Ubuf + fN1;
    size_t h2cap = fN0 + fN1;
    float* H2buf = ws; // alias of X0b+X1f region (both dead by then)
    _Float16* WhBase = (_Float16*)(ACC0 + fN0);
    _Float16* Wn_h   = WhBase;
    _Float16* We_h   = WhBase + 16384;
    _Float16* Wt_h   = WhBase + 32768;
    _Float16* Wtri_h = WhBase + 49152;
    (void)h2cap; (void)ws_size; (void)n_in; (void)out_size;

    const int TB = 256;

    // ---- weight conversions (tiny) ----
    cvt_f16_kernel<<<cdiv(16384, TB), TB, 0, stream>>>(Wn,   Wn_h,   16384);
    cvt_f16_kernel<<<cdiv(16384, TB), TB, 0, stream>>>(We,   We_h,   16384);
    cvt_f16_kernel<<<cdiv(16384, TB), TB, 0, stream>>>(Wt,   Wt_h,   16384);
    cvt_f16_kernel<<<cdiv(16384, TB), TB, 0, stream>>>(Wtri, Wtri_h, 16384);

    // ---- features ----
    binarize_kernel<<<cdiv(fN0, TB), TB, 0, stream>>>(X0, X0b, (int)fN0);
    edge_feat_kernel<<<cdiv((long long)N1 * 32, TB), TB, 0, stream>>>(eidx, X0b, X1f, N1);
    tri_feat_kernel<<<cdiv((long long)N2 * 32, TB), TB, 0, stream>>>(tidx, X0b, X2f, N2);

    dim3 gblk(32, 4); // 4 waves/block, each wave -> 16 rows x 128 cols

    // ---- H0 = prelu(L0 @ X0b @ Wn^T + bn)  -> ACC0 ----
    hipMemsetAsync(Sbuf, 0, fN0 * sizeof(float), stream);
    spmm_atomic_kernel<<<cdiv((long long)nnz0 * 32, TB), TB, 0, stream>>>(L0r, L0c, L0v, X0b, Sbuf, nnz0);
    gemm128_kernel<true, false><<<cdiv(N0, 64), gblk, 0, stream>>>(Sbuf, Wn_h, bn, prelu_w, ACC0, N0);

    // ---- H1 = prelu(L1 @ X1f @ We^T + be)  -> U ----
    hipMemsetAsync(Sbuf, 0, fN1 * sizeof(float), stream);
    spmm_atomic_kernel<<<cdiv((long long)nnz1 * 32, TB), TB, 0, stream>>>(L1r, L1c, L1v, X1f, Sbuf, nnz1);
    gemm128_kernel<true, false><<<cdiv(N1, 64), gblk, 0, stream>>>(Sbuf, We_h, be, prelu_w, Ubuf, N1);

    // ---- H2 = prelu(L2 @ X2f @ Wt^T + bt)  -> H2buf (aliases dead X0b/X1f) ----
    hipMemsetAsync(Sbuf, 0, fN2 * sizeof(float), stream);
    spmm_atomic_kernel<<<cdiv((long long)nnz2 * 32, TB), TB, 0, stream>>>(L2r, L2c, L2v, X2f, Sbuf, nnz2);
    gemm128_kernel<true, false><<<cdiv(N2, 64), gblk, 0, stream>>>(Sbuf, Wt_h, bt, prelu_w, H2buf, N2);

    // ---- T = (B2 @ H2) @ Wtri^T + btri ; U = H1 + T  (fused accumulate GEMM) ----
    hipMemsetAsync(Sbuf, 0, fN1 * sizeof(float), stream);
    spmm_atomic_kernel<<<cdiv((long long)nnzb2 * 32, TB), TB, 0, stream>>>(B2r, B2c, B2v, H2buf, Sbuf, nnzb2);
    gemm128_kernel<false, true><<<cdiv(N1, 64), gblk, 0, stream>>>(Sbuf, Wtri_h, btri, prelu_w, Ubuf, N1);

    // ---- ACC0 += B1 @ U   (== spmm(B1,H1) + spmm(B1,T)) ----
    spmm_atomic_kernel<<<cdiv((long long)nnzb1 * 32, TB), TB, 0, stream>>>(B1r, B1c, B1v, Ubuf, ACC0, nnzb1);

    // ---- out = ACC0 / 3 ----
    scale_out_kernel<<<cdiv(fN0, TB), TB, 0, stream>>>(ACC0, (float*)d_out, (int)fN0);
}
